// SiglipAttention_35648228557483
// MI455X (gfx1250) — compile-verified, matrix operand order checked by reading, not run on hardware
//
#include <hip/hip_runtime.h>
#include <hip/hip_bf16.h>

// ---------------------------------------------------------------------------
// SiglipAttention forward for MI455X (gfx1250, wave32, WMMA).
// B=8, S=1024, D=1152, H=16, HD=72.
// All GEMMs: bf16 WMMA, fp32 accumulate (v_wmma_f32_16x16x32_bf16).
// Shared operand tiles staged into LDS with global_load_async_to_lds_b128
// (ASYNCcnt), double-buffered; softmax in fp32.
// ---------------------------------------------------------------------------

#define BB 8
#define SS 1024
#define DD 1152
#define HH 16
#define HD 72
#define HDP 96                 // HD padded to multiple of 32 for WMMA K loops
#define MROWS (BB * SS)        // 8192
#define SCALE 0.117851130197758f   // 72^-0.5

typedef __bf16 bf16x8  __attribute__((ext_vector_type(8)));
typedef __bf16 bf16x16 __attribute__((ext_vector_type(16)));
typedef float  f32x8   __attribute__((ext_vector_type(8)));
typedef float  v8f     __attribute__((ext_vector_type(8)));

// ---- WMMA wrapper ----------------------------------------------------------
static __device__ __forceinline__ v8f wmma_bf16(bf16x16 a, bf16x16 b, v8f c) {
  return __builtin_amdgcn_wmma_f32_16x16x32_bf16(false, a, false, b,
                                                 (short)0, c, false, false);
}

// ---- CDNA5 async copy global -> LDS (ASYNCcnt) -----------------------------
// LDS address of a __shared__ object is addr[31:0] of its generic pointer
// (ISA 10.2: LDS aperture occupies the high 32 bits only).
static __device__ __forceinline__ void async_cp_b128(const __bf16* g,
                                                     const __bf16* l) {
  unsigned lds = (unsigned)(size_t)(const void*)l;
  unsigned long long ga = (unsigned long long)(size_t)(const void*)g;
  asm volatile("global_load_async_to_lds_b128 %0, %1, off"
               :: "v"(lds), "v"(ga) : "memory");
}
static __device__ __forceinline__ void async_wait_all() {
  asm volatile("s_wait_asynccnt 0" ::: "memory");
}

// Stage nrows rows of `rowHalfs` bf16 each (rowHalfs % 8 == 0) from global
// (row stride gstride) into LDS (row stride rowHalfs), one b128 per thread.
static __device__ __forceinline__ void stage_async(const __bf16* __restrict__ g,
                                                   int gstride,
                                                   __bf16* __restrict__ s,
                                                   int nrows, int rowHalfs) {
  const int cpr = rowHalfs >> 3;                 // b128 chunks per row
  for (int idx = threadIdx.x; idx < nrows * cpr; idx += blockDim.x) {
    const int row = idx / cpr;
    const int c = (idx - row * cpr) << 3;
    async_cp_b128(g + (size_t)row * gstride + c, s + row * rowHalfs + c);
  }
}

// ---- Operand tile loaders --------------------------------------------------
// 16x32 bf16 operand tile from a row-major [16+, K] matrix (NT layout): per
// documented 16-bit A layout, lanes 0-15 hold K chunks {0..7,16..23}, lanes
// 16-31 hold {8..15,24..31}. Works on global or LDS (__shared__) pointers.
static __device__ __forceinline__ bf16x16 ld_tile_bf16(const __bf16* __restrict__ p,
                                                       int stride) {
  const int lane = threadIdx.x & 31;
  const int r    = lane & 15;
  const int kh   = (lane >> 4) << 3;   // 0 or 8
  const __bf16* base = p + (size_t)r * stride + kh;
  union { bf16x16 v; bf16x8 h[2]; } u;
  u.h[0] = *(const bf16x8*)(base);
  u.h[1] = *(const bf16x8*)(base + 16);
  return u.v;
}

// Same pattern, fp32 source (attn probabilities) converted to bf16 in-reg.
static __device__ __forceinline__ bf16x16 ld_tile_f32_as_bf16(const float* __restrict__ p,
                                                              int stride) {
  const int lane = threadIdx.x & 31;
  const int r    = lane & 15;
  const int kh   = (lane >> 4) << 3;
  const float* base = p + (size_t)r * stride + kh;
  f32x8 lo = *(const f32x8*)(base);
  f32x8 hi = *(const f32x8*)(base + 16);
  union { bf16x16 v; bf16x8 h[2]; } u;
  u.h[0] = __builtin_convertvector(lo, bf16x8);
  u.h[1] = __builtin_convertvector(hi, bf16x8);
  return u.v;
}

// ---- Elementwise helpers ---------------------------------------------------
__global__ void cast_f32_to_bf16_k(const float* __restrict__ src,
                                   __bf16* __restrict__ dst, size_t n) {
  size_t i = (size_t)blockIdx.x * blockDim.x + threadIdx.x;
  if (i < n) dst[i] = (__bf16)src[i];
}

__global__ void zero_b128_k(float4* __restrict__ p, size_t n16) {
  size_t i = (size_t)blockIdx.x * blockDim.x + threadIdx.x;
  if (i < n16) p[i] = make_float4(0.f, 0.f, 0.f, 0.f);
}

// ---- NT GEMM: C[M,N] = A[M,K] * W[N,K]^T + bias ----------------------------
// 256-thread block = 8 waves. Block tile: 256 rows x 64 cols; each wave owns
// a 32x64 tile (8 WMMAs per K-step). The 64x32 weight tile is async-staged
// into LDS (double-buffered) and shared by all waves. mode selects scatter:
//   0: Q -> bf16 [B,H,S,HDP] * SCALE   1: K -> bf16 [B,H,S,HDP]
//   2: V -> bf16 [B,H,HDP,S]           3: fp32 row-major [M,N] (d_out)
__global__ void gemm_nt_proj_k(const __bf16* __restrict__ A,
                               const __bf16* __restrict__ W,
                               const float* __restrict__ bias,
                               void* __restrict__ dst,
                               int K, int mode, float scale) {
  __shared__ __bf16 sB[2][64 * 32];
  const int wave = threadIdx.x >> 5;
  const int m0 = blockIdx.x * 256 + wave * 32;   // this wave's first row
  const int n0 = blockIdx.y * 64;

  v8f acc[2][4] = {};
  const __bf16* Ab = A + (size_t)m0 * K;
  const __bf16* Wb = W + (size_t)n0 * K;

  stage_async(Wb, K, sB[0], 64, 32);             // prologue: stage kk=0
  async_wait_all();
  __syncthreads();

  for (int kk = 0; kk < K; kk += 32) {
    const int cur = (kk >> 5) & 1;
    if (kk + 32 < K) {
      stage_async(Wb + kk + 32, K, sB[cur ^ 1], 64, 32);
      __builtin_prefetch(Ab + ((threadIdx.x & 15) * (size_t)K) + kk + 32, 0, 1);
    }
    bf16x16 a0 = ld_tile_bf16(Ab + kk, K);
    bf16x16 a1 = ld_tile_bf16(Ab + (size_t)16 * K + kk, K);
#pragma unroll
    for (int t = 0; t < 4; ++t) {
      bf16x16 b = ld_tile_bf16(&sB[cur][t * 16 * 32], 32);
      acc[0][t] = wmma_bf16(a0, b, acc[0][t]);
      acc[1][t] = wmma_bf16(a1, b, acc[1][t]);
    }
    async_wait_all();
    __syncthreads();
  }

  const int lane  = threadIdx.x & 31;
  const int nin   = lane & 15;
  const int mbase = (lane >> 4) << 3;
#pragma unroll
  for (int mt = 0; mt < 2; ++mt) {
#pragma unroll
    for (int t = 0; t < 4; ++t) {
      const int gn = n0 + t * 16 + nin;
      const float bv = bias[gn];
      union { v8f v; float e[8]; } u; u.v = acc[mt][t];
#pragma unroll
      for (int r = 0; r < 8; ++r) {
        const int gm = m0 + mt * 16 + mbase + r;
        const float val = (u.e[r] + bv) * scale;
        if (mode == 3) {
          ((float*)dst)[(size_t)gm * DD + gn] = val;
        } else {
          const int b = gm >> 10;          // S == 1024
          const int s = gm & (SS - 1);
          const int h = gn / HD;
          const int hd = gn - h * HD;
          __bf16* db = (__bf16*)dst;
          if (mode == 2)   // V transposed: [B,H,HDP,S]
            db[(((size_t)(b * HH + h) * HDP + hd) << 10) + s] = (__bf16)val;
          else             // Q/K: [B,H,S,HDP]
            db[(((size_t)(b * HH + h) << 10) + s) * HDP + hd] = (__bf16)val;
        }
      }
    }
  }
}

// ---- Scores: attn_raw[bh,q,k] = (Q*scale) . K  (K dim padded to 96) --------
// Block: 256 q-rows x 64 keys; the 64x96 key tile is async-staged to LDS once
// and reused for all 3 K-steps by all 8 waves (each wave: 32 q-rows).
__global__ void attn_scores_k(const __bf16* __restrict__ Qh,
                              const __bf16* __restrict__ Kh,
                              float* __restrict__ attn) {
  __shared__ __bf16 sK[64 * HDP];                // 12 KB
  const int bh = blockIdx.z;
  const int wave = threadIdx.x >> 5;
  const int q0 = blockIdx.x * 256 + wave * 32;
  const int k0 = blockIdx.y * 64;
  const __bf16* Q = Qh + (size_t)bh * SS * HDP + (size_t)q0 * HDP;
  const __bf16* Kp = Kh + (size_t)bh * SS * HDP + (size_t)k0 * HDP;

  stage_async(Kp, HDP, sK, 64, HDP);
  async_wait_all();
  __syncthreads();

  v8f acc[2][4] = {};
#pragma unroll
  for (int kk = 0; kk < HDP; kk += 32) {
    bf16x16 a0 = ld_tile_bf16(Q + kk, HDP);
    bf16x16 a1 = ld_tile_bf16(Q + (size_t)16 * HDP + kk, HDP);
#pragma unroll
    for (int t = 0; t < 4; ++t) {
      bf16x16 b = ld_tile_bf16(&sK[t * 16 * HDP + kk], HDP);
      acc[0][t] = wmma_bf16(a0, b, acc[0][t]);
      acc[1][t] = wmma_bf16(a1, b, acc[1][t]);
    }
  }

  const int lane  = threadIdx.x & 31;
  const int nin   = lane & 15;
  const int mbase = (lane >> 4) << 3;
  float* orow = attn + (size_t)bh * SS * SS;
#pragma unroll
  for (int mt = 0; mt < 2; ++mt) {
#pragma unroll
    for (int t = 0; t < 4; ++t) {
      union { v8f v; float e[8]; } u; u.v = acc[mt][t];
#pragma unroll
      for (int r = 0; r < 8; ++r)
        orow[(size_t)(q0 + mt * 16 + mbase + r) * SS + (k0 + t * 16 + nin)] = u.e[r];
    }
  }
}

// ---- Row softmax over 1024 columns, fp32, one wave per row -----------------
__global__ void softmax_rows_k(float* __restrict__ attn) {
  const int wid  = threadIdx.x >> 5;
  const int lane = threadIdx.x & 31;
  const size_t row = (size_t)blockIdx.x * (blockDim.x >> 5) + wid;
  float* p = attn + row * SS;

  float v[32];
  float m = -3.402823466e38f;
#pragma unroll
  for (int i = 0; i < 32; ++i) { v[i] = p[lane + i * 32]; m = fmaxf(m, v[i]); }
#pragma unroll
  for (int off = 16; off > 0; off >>= 1) m = fmaxf(m, __shfl_xor(m, off, 32));
  float s = 0.f;
#pragma unroll
  for (int i = 0; i < 32; ++i) { v[i] = __expf(v[i] - m); s += v[i]; }
#pragma unroll
  for (int off = 16; off > 0; off >>= 1) s += __shfl_xor(s, off, 32);
  const float inv = 1.0f / s;
#pragma unroll
  for (int i = 0; i < 32; ++i) p[lane + i * 32] = v[i] * inv;
}

// ---- attn @ V: AO[b*S+q, h*72+hd] = sum_k attn[bh,q,k] * Vt[bh,hd,k] -------
// Block: 256 q-rows; wave: 32 q-rows x 80 head-dims (10 WMMAs per K-step).
// Per-K-step 80x32 Vt tile async-staged to LDS, double-buffered, shared.
__global__ void attn_v_k(const float* __restrict__ attn,
                         const __bf16* __restrict__ Vt,
                         __bf16* __restrict__ AO) {
  __shared__ __bf16 sV[2][80 * 32];              // 2 x 5 KB
  const int bh = blockIdx.y;
  const int wave = threadIdx.x >> 5;
  const int q0 = blockIdx.x * 256 + wave * 32;
  const int b = bh >> 4, h = bh & 15;
  const float* Ar = attn + (size_t)bh * SS * SS + (size_t)q0 * SS;
  const __bf16* Vb = Vt + (size_t)bh * HDP * SS;

  v8f acc[2][5] = {};
  stage_async(Vb, SS, sV[0], 80, 32);
  async_wait_all();
  __syncthreads();

  for (int kk = 0; kk < SS; kk += 32) {
    const int cur = (kk >> 5) & 1;
    if (kk + 32 < SS) {
      stage_async(Vb + kk + 32, SS, sV[cur ^ 1], 80, 32);
      __builtin_prefetch(Ar + ((threadIdx.x & 15) * (size_t)SS) + kk + 32, 0, 1);
    }
    bf16x16 a0 = ld_tile_f32_as_bf16(Ar + kk, SS);
    bf16x16 a1 = ld_tile_f32_as_bf16(Ar + (size_t)16 * SS + kk, SS);
#pragma unroll
    for (int t = 0; t < 5; ++t) {
      bf16x16 bt = ld_tile_bf16(&sV[cur][t * 16 * 32], 32);
      acc[0][t] = wmma_bf16(a0, bt, acc[0][t]);
      acc[1][t] = wmma_bf16(a1, bt, acc[1][t]);
    }
    async_wait_all();
    __syncthreads();
  }

  const int lane  = threadIdx.x & 31;
  const int nin   = lane & 15;
  const int mbase = (lane >> 4) << 3;
#pragma unroll
  for (int mt = 0; mt < 2; ++mt) {
#pragma unroll
    for (int t = 0; t < 5; ++t) {
      const int gn = t * 16 + nin;          // head-dim column, 0..79
      if (gn >= HD) continue;
      union { v8f v; float e[8]; } u; u.v = acc[mt][t];
#pragma unroll
      for (int r = 0; r < 8; ++r) {
        const size_t grow = ((size_t)b << 10) + q0 + mt * 16 + mbase + r;
        AO[grow * DD + h * HD + gn] = (__bf16)u.e[r];
      }
    }
  }
}

// ---------------------------------------------------------------------------
extern "C" void kernel_launch(void* const* d_in, const int* in_sizes, int n_in,
                              void* d_out, int out_size, void* d_ws, size_t ws_size,
                              hipStream_t stream) {
  const float* X  = (const float*)d_in[0];
  const float* Wq = (const float*)d_in[1];
  const float* bq = (const float*)d_in[2];
  const float* Wk = (const float*)d_in[3];
  const float* bk = (const float*)d_in[4];
  const float* Wv = (const float*)d_in[5];
  const float* bv = (const float*)d_in[6];
  const float* Wo = (const float*)d_in[7];
  const float* bo = (const float*)d_in[8];

  float* out  = (float*)d_out;
  float* attn = out + (size_t)BB * SS * DD;   // tuple member 2 of d_out

  // Workspace carve-up (bf16 staging buffers), ~124 MB total.
  char* ws = (char*)d_ws;
  size_t off = 0;
  const size_t XB_BYTES = (size_t)MROWS * DD * 2;          // 18.9 MB
  const size_t W_BYTES  = (size_t)DD * DD * 2;             // 2.65 MB each
  const size_t HQ_BYTES = (size_t)BB * HH * SS * HDP * 2;  // 25.2 MB each

  __bf16* Xb  = (__bf16*)(ws + off); off += XB_BYTES;
  __bf16* Wqb = (__bf16*)(ws + off); off += W_BYTES;
  __bf16* Wkb = (__bf16*)(ws + off); off += W_BYTES;
  __bf16* Wvb = (__bf16*)(ws + off); off += W_BYTES;
  __bf16* Wob = (__bf16*)(ws + off); off += W_BYTES;
  __bf16* Qh  = (__bf16*)(ws + off); off += HQ_BYTES;
  __bf16* Kh  = (__bf16*)(ws + off); off += HQ_BYTES;
  __bf16* Vt  = (__bf16*)(ws + off); off += HQ_BYTES;
  __bf16* AO  = (__bf16*)(ws + off); off += XB_BYTES;

  const int T = 256;
  // 1) fp32 -> bf16 casts
  {
    size_t n = (size_t)MROWS * DD;
    cast_f32_to_bf16_k<<<(unsigned)((n + T - 1) / T), T, 0, stream>>>(X, Xb, n);
    size_t nw = (size_t)DD * DD;
    unsigned gw = (unsigned)((nw + T - 1) / T);
    cast_f32_to_bf16_k<<<gw, T, 0, stream>>>(Wq, Wqb, nw);
    cast_f32_to_bf16_k<<<gw, T, 0, stream>>>(Wk, Wkb, nw);
    cast_f32_to_bf16_k<<<gw, T, 0, stream>>>(Wv, Wvb, nw);
    cast_f32_to_bf16_k<<<gw, T, 0, stream>>>(Wo, Wob, nw);
  }
  // 2) zero padded head buffers (pads hd 72..95 must be exactly 0)
  {
    size_t n16 = HQ_BYTES / 16;
    unsigned g = (unsigned)((n16 + T - 1) / T);
    zero_b128_k<<<g, T, 0, stream>>>((float4*)Qh, n16);
    zero_b128_k<<<g, T, 0, stream>>>((float4*)Kh, n16);
    zero_b128_k<<<g, T, 0, stream>>>((float4*)Vt, n16);
  }
  // 3) Q/K/V projections (bias folded in; Q pre-scaled by 1/sqrt(HD))
  {
    dim3 g(MROWS / 256, DD / 64, 1);
    gemm_nt_proj_k<<<g, 256, 0, stream>>>(Xb, Wqb, bq, (void*)Qh, DD, 0, SCALE);
    gemm_nt_proj_k<<<g, 256, 0, stream>>>(Xb, Wkb, bk, (void*)Kh, DD, 1, 1.0f);
    gemm_nt_proj_k<<<g, 256, 0, stream>>>(Xb, Wvb, bv, (void*)Vt, DD, 2, 1.0f);
  }
  // 4) scores -> attn region of d_out (fp32)
  {
    dim3 g(SS / 256, SS / 64, BB * HH);
    attn_scores_k<<<g, 256, 0, stream>>>(Qh, Kh, attn);
  }
  // 5) softmax in place (131072 rows, 8 rows per 256-thread block)
  {
    unsigned rows = BB * HH * SS;
    softmax_rows_k<<<rows / 8, 256, 0, stream>>>(attn);
  }
  // 6) attn @ V -> AO (bf16 [8192,1152])
  {
    dim3 g(SS / 256, BB * HH, 1);
    attn_v_k<<<g, 256, 0, stream>>>(attn, Vt, AO);
  }
  // 7) final projection -> d_out fp32
  {
    dim3 g(MROWS / 256, DD / 64, 1);
    gemm_nt_proj_k<<<g, 256, 0, stream>>>(AO, Wob, bo, (void*)out, DD, 3, 1.0f);
  }
}
